// RNNDecoder_32633161515022
// MI455X (gfx1250) — matrix operand order repaired and checked by reading, not compiled, not run on hardware
//
#include <hip/hip_runtime.h>

typedef __bf16 bf16;
typedef __attribute__((ext_vector_type(16))) __bf16 v16bf;
typedef __attribute__((ext_vector_type(8)))  __bf16 v8bf;
typedef __attribute__((ext_vector_type(8)))  float  v8f;

#define T_STEPS 127
#define BATCH   64
#define SEQ     256
#define HID     1024
#define EMB     512
#define VOCAB   32000
#define KX      2560   // [o_prev(1024) | Y_t(512) | h(1024)]
#define KU      3072   // [a_t(2048) | h(1024)]
#define PAD_ID  0

// ---------- bf16 helpers (bit-exact, no reliance on __bf16 conversions) ----------
__device__ __forceinline__ bf16 f2bf(float x) {
    unsigned u; __builtin_memcpy(&u, &x, 4);
    unsigned r = u + 0x7FFFu + ((u >> 16) & 1u);
    unsigned short h = (unsigned short)(r >> 16);
    bf16 o; __builtin_memcpy(&o, &h, 2);
    return o;
}
__device__ __forceinline__ float bf2f(bf16 x) {
    unsigned short h; __builtin_memcpy(&h, &x, 2);
    unsigned u = ((unsigned)h) << 16;
    float f; __builtin_memcpy(&f, &u, 4);
    return f;
}

// ---------- WMMA fragment loaders (CDNA5 16x16x32 bf16 layouts, wave32) ----------
// A 16x32 (MxK), row-major source. Lane L: m=L%16, half=L/16.
// elems 0..15 -> K = 16*half + 0..15
__device__ __forceinline__ v16bf load_a16x32(const bf16* A, int lda) {
    int lane = threadIdx.x & 31;
    int m = lane & 15, hf = lane >> 4;
    const bf16* p = A + (long)m * lda + 16 * hf;
    v8bf lo = *(const v8bf*)p;
    v8bf hi = *(const v8bf*)(p + 8);
    v16bf r;
#pragma unroll
    for (int i = 0; i < 8; i++) { r[i] = lo[i]; r[i + 8] = hi[i]; }
    return r;
}
// B 32x16 (KxN) taken from row-major W (N x K): B[k][n] = W[n0+n][k0+k].
// Lane L: n=L%16, half=L/16, elems j -> k = 16*half + j (contiguous 32B load).
__device__ __forceinline__ v16bf load_b32x16(const bf16* W, long n0, int ldw, int k0) {
    int lane = threadIdx.x & 31;
    return *(const v16bf*)(W + (n0 + (lane & 15)) * (long)ldw + k0 + 16 * (lane >> 4));
}
__device__ __forceinline__ v8f wmma_bf16(v16bf a, v16bf b, v8f c) {
    return __builtin_amdgcn_wmma_f32_16x16x32_bf16(false, a, false, b, (short)0, c, false, false);
}
// C/D: elem r of v8f -> m = r + 8*(lane/16), n = lane%16

// ---------------- prep kernels ----------------
__global__ void k_f2bf(const float* __restrict__ src, bf16* __restrict__ dst, long n) {
    long i = (long)blockIdx.x * blockDim.x + threadIdx.x;
    if (i < n) dst[i] = f2bf(src[i]);
}

__global__ void k_wcomb(const float* __restrict__ Wih, const float* __restrict__ Whh,
                        bf16* __restrict__ W, long n) {
    long i = (long)blockIdx.x * blockDim.x + threadIdx.x;
    if (i >= n) return;
    long r = i / KX; int k = (int)(i % KX);
    float v = (k < 1536) ? Wih[r * 1536 + k] : Whh[r * 1024 + (k - 1536)];
    W[i] = f2bf(v);
}

__global__ void k_embed(const int* __restrict__ ids, const float* __restrict__ Et,
                        bf16* __restrict__ Y, long n) {
    long i = (long)blockIdx.x * blockDim.x + threadIdx.x;
    if (i >= n) return;
    int e = (int)(i % EMB); long tb = i / EMB;      // t*64+b, t in [0,126] = ids[:-1]
    int id = ids[tb];
    Y[i] = f2bf(Et[(long)id * EMB + e]);
}

__global__ void k_init(const float* __restrict__ h0, const float* __restrict__ c0,
                       const bf16* __restrict__ Y, bf16* __restrict__ X,
                       float* __restrict__ hf, float* __restrict__ cf) {
    int i = blockIdx.x * blockDim.x + threadIdx.x;
    if (i < BATCH * KX) {
        int b = i / KX, k = i % KX;
        bf16 v;
        if (k < 1024) { unsigned short z = 0; __builtin_memcpy(&v, &z, 2); }   // o_init = 0
        else if (k < 1536) v = Y[b * EMB + (k - 1024)];                         // Y_0
        else v = f2bf(h0[b * HID + (k - 1536)]);                                // h0
        X[i] = v;
    }
    if (i < BATCH * HID) { hf[i] = h0[i]; cf[i] = c0[i]; }
}

// ---------------- enc_proj: (16384 x 2048) x (2048 x 1024) + b_att -> bf16 ----------------
// grid (1024, 8), 128 thr; each wave: 2 adjacent n-tiles sharing the A fragment
__global__ void k_encproj(const bf16* __restrict__ Avc, const bf16* __restrict__ Watt,
                          const float* __restrict__ batt, bf16* __restrict__ encp) {
    int wave = threadIdx.x >> 5;
    long m0 = (long)blockIdx.x * 16;
    long n0 = blockIdx.y * 128 + wave * 32;
    v8f acc0 = {}, acc1 = {};
    for (int kk = 0; kk < 2 * HID; kk += 32) {
        v16bf a  = load_a16x32(Avc + m0 * (2 * HID) + kk, 2 * HID);
        v16bf b0 = load_b32x16(Watt, n0,      2 * HID, kk);
        v16bf b1 = load_b32x16(Watt, n0 + 16, 2 * HID, kk);
        acc0 = wmma_bf16(a, b0, acc0);
        acc1 = wmma_bf16(a, b1, acc1);
    }
    int lane = threadIdx.x & 31;
    int n = lane & 15, hf = lane >> 4;
    float bias0 = batt[n0 + n], bias1 = batt[n0 + 16 + n];
#pragma unroll
    for (int r = 0; r < 8; r++) {
        long m = m0 + r + 8 * hf;
        encp[m * HID + n0 + n]      = f2bf(acc0[r] + bias0);
        encp[m * HID + n0 + 16 + n] = f2bf(acc1[r] + bias1);
    }
}

// ---------------- step1: gates GEMM (64x2560)x(2560->4096) + LSTM pointwise ----------------
// grid (4 batch-tiles, 32 j-tiles of 32), 128 thr (wave w = gate w, 2 j-tiles each)
__global__ void k_step1(const bf16* __restrict__ X, const bf16* __restrict__ Wc,
                        float* __restrict__ cf, float* __restrict__ hf, bf16* Xn) {
    __shared__ float g4[4][16][32];
    int wave = threadIdx.x >> 5, lane = threadIdx.x & 31;
    int b0 = blockIdx.x * 16;
    int j0 = blockIdx.y * 32;
    long n0 = (long)wave * 1024 + j0;
    v8f acc0 = {}, acc1 = {};
    for (int kk = 0; kk < KX; kk += 32) {
        v16bf a  = load_a16x32(X + (long)b0 * KX + kk, KX);
        v16bf b0v = load_b32x16(Wc, n0,      KX, kk);
        v16bf b1v = load_b32x16(Wc, n0 + 16, KX, kk);
        acc0 = wmma_bf16(a, b0v, acc0);
        acc1 = wmma_bf16(a, b1v, acc1);
    }
    int n = lane & 15, hv = lane >> 4;
#pragma unroll
    for (int r = 0; r < 8; r++) {
        g4[wave][r + 8 * hv][n]      = acc0[r];
        g4[wave][r + 8 * hv][n + 16] = acc1[r];
    }
    __syncthreads();
    for (int e = threadIdx.x; e < 512; e += 128) {
        int m = e >> 5, nn = e & 31;
        int b = b0 + m, j = j0 + nn;
        float gi = g4[0][m][nn], gfv = g4[1][m][nn], gg = g4[2][m][nn], go = g4[3][m][nn];
        gi  = 1.f / (1.f + __expf(-gi));
        gfv = 1.f / (1.f + __expf(-gfv));
        gg  = tanhf(gg);
        go  = 1.f / (1.f + __expf(-go));
        float c = gfv * cf[b * HID + j] + gi * gg;
        float hh = go * tanhf(c);
        cf[b * HID + j] = c;
        hf[b * HID + j] = hh;
        Xn[(long)b * KX + 1536 + j] = f2bf(hh);     // h into next-step X
    }
}

// ---------------- step2: attention (scores, softmax, context) + build Xu=[a_t|h] ----------------
// one block per batch, 256 threads (= SEQ)
__global__ void k_step2(const float* __restrict__ hf, const bf16* __restrict__ encp,
                        const bf16* __restrict__ vc, bf16* __restrict__ Xu) {
    __shared__ float hs[HID];
    __shared__ float alpha[SEQ];
    __shared__ float red[256];
    int b = blockIdx.x, tid = threadIdx.x;
    for (int k = tid; k < HID; k += 256) hs[k] = hf[b * HID + k];
    __syncthreads();
    const bf16* erow = encp + ((long)b * SEQ + tid) * HID;
    float e = 0.f;
    for (int k = 0; k < HID; k++) e += bf2f(erow[k]) * hs[k];
    red[tid] = e; __syncthreads();
    for (int s = 128; s > 0; s >>= 1) { if (tid < s) red[tid] = fmaxf(red[tid], red[tid + s]); __syncthreads(); }
    float mx = red[0]; __syncthreads();
    float ex = __expf(e - mx);
    red[tid] = ex; __syncthreads();
    for (int s = 128; s > 0; s >>= 1) { if (tid < s) red[tid] += red[tid + s]; __syncthreads(); }
    float inv = 1.f / red[0];
    alpha[tid] = ex * inv;
    __syncthreads();
    float acc[8];
#pragma unroll
    for (int i = 0; i < 8; i++) acc[i] = 0.f;
    for (int s = 0; s < SEQ; s++) {
        float al = alpha[s];
        const bf16* vrow = vc + ((long)b * SEQ + s) * (2 * HID);
#pragma unroll
        for (int i = 0; i < 8; i++) acc[i] += al * bf2f(vrow[tid + i * 256]);
    }
#pragma unroll
    for (int i = 0; i < 8; i++) Xu[(long)b * KU + tid + i * 256] = f2bf(acc[i]);
    for (int k = tid; k < HID; k += 256) Xu[(long)b * KU + 2048 + k] = f2bf(hs[k]);
}

// ---------------- step3: o_t = tanh(Xu @ W_u^T); write o into X_next + combined; stage Y_{t+1} ----------------
// grid (4 batch-tiles, 8 n-tiles of 128), 128 thr (4 waves, 2 n-tiles each)
__global__ void k_step3(const bf16* __restrict__ Xu, const bf16* __restrict__ Wu,
                        bf16* Xn, bf16* __restrict__ comb, const bf16* __restrict__ Y, int t) {
    int wave = threadIdx.x >> 5, lane = threadIdx.x & 31;
    int b0 = blockIdx.x * 16;
    long n0 = blockIdx.y * 128 + wave * 32;
    v8f acc0 = {}, acc1 = {};
    for (int kk = 0; kk < KU; kk += 32) {
        v16bf a  = load_a16x32(Xu + (long)b0 * KU + kk, KU);
        v16bf b0v = load_b32x16(Wu, n0,      KU, kk);
        v16bf b1v = load_b32x16(Wu, n0 + 16, KU, kk);
        acc0 = wmma_bf16(a, b0v, acc0);
        acc1 = wmma_bf16(a, b1v, acc1);
    }
    int n = lane & 15, hv = lane >> 4;
#pragma unroll
    for (int r = 0; r < 8; r++) {
        int b = b0 + r + 8 * hv;
        bf16 v0 = f2bf(tanhf(acc0[r]));
        bf16 v1 = f2bf(tanhf(acc1[r]));
        Xn[(long)b * KX + n0 + n]      = v0;                          // o_prev for next step
        Xn[(long)b * KX + n0 + 16 + n] = v1;
        comb[((long)t * BATCH + b) * HID + n0 + n]      = v0;         // combined[t] for vocab GEMM
        comb[((long)t * BATCH + b) * HID + n0 + 16 + n] = v1;
    }
    if (blockIdx.y == 0 && t + 1 < T_STEPS) {                        // stage Y_{t+1}
        for (int e = threadIdx.x; e < 16 * EMB; e += 128) {
            int m = e >> 9, k = e & 511;
            int b = b0 + m;
            Xn[(long)b * KX + 1024 + k] = Y[((long)(t + 1) * BATCH + b) * EMB + k];
        }
    }
}

// ---------------- vocab scoring: online logsumexp over V, capture gold logit ----------------
// grid 508 blocks (16-row tiles of 8128 rows), 256 thr = 8 waves
// each wave processes PAIRS of 16-col tiles (shared A fragment, dual WMMA chains):
// 2000 tiles -> 1000 pairs -> 125 iterations per wave (exact)
__global__ void k_score(const bf16* __restrict__ comb, const bf16* __restrict__ Wv,
                        const float* __restrict__ bvoc, const int* __restrict__ ids,
                        float* __restrict__ glp) {
    __shared__ bf16 As[16 * HID];
    __shared__ float rmax[8][16][16];
    __shared__ float rsum[8][16][16];
    __shared__ float gval[16];
    __shared__ int   gcol[16];
    int tid = threadIdx.x;
    int wave = tid >> 5, lane = tid & 31;
    int m0 = blockIdx.x * 16;          // flat row = t*64+b; 16 | 64 so all rows share t
    int tt = m0 >> 6;
    for (int i = tid; i < 16 * HID; i += 256) As[i] = comb[(long)m0 * HID + i];
    if (tid < 16) {
        int b = (m0 & 63) + tid;
        gcol[tid] = ids[(long)(tt + 1) * BATCH + b];   // gold = target_ids[t+1][b]
        gval[tid] = 0.f;
    }
    __syncthreads();
    float mx8[8], sm8[8];
#pragma unroll
    for (int r = 0; r < 8; r++) { mx8[r] = -1e30f; sm8[r] = 0.f; }
    int n = lane & 15, hv = lane >> 4;
    for (int ti = wave * 2; ti < VOCAB / 16; ti += 16) {
        long n0 = (long)ti * 16;
        // prefetch next pair's 32 B-rows for this wave (L2 -> WGP pull overlaps compute)
        if (ti + 16 < VOCAB / 16)
            __builtin_prefetch(Wv + (n0 + 256 + lane) * (long)HID, 0, 1);
        v8f acc0 = {}, acc1 = {};
        for (int kk = 0; kk < HID; kk += 32) {
            v16bf a  = load_a16x32((const bf16*)As + kk, HID);
            v16bf b0 = load_b32x16(Wv, n0,      HID, kk);
            v16bf b1 = load_b32x16(Wv, n0 + 16, HID, kk);
            acc0 = wmma_bf16(a, b0, acc0);
            acc1 = wmma_bf16(a, b1, acc1);
        }
        float bias0 = bvoc[n0 + n];
        float bias1 = bvoc[n0 + 16 + n];
#pragma unroll
        for (int r = 0; r < 8; r++) {
            int m = r + 8 * hv;
            float v0 = acc0[r] + bias0;
            float v1 = acc1[r] + bias1;
            if (gcol[m] == (int)n0 + n)      gval[m] = v0;
            if (gcol[m] == (int)n0 + 16 + n) gval[m] = v1;
            if (v0 > mx8[r]) { sm8[r] = sm8[r] * __expf(mx8[r] - v0) + 1.f; mx8[r] = v0; }
            else             { sm8[r] += __expf(v0 - mx8[r]); }
            if (v1 > mx8[r]) { sm8[r] = sm8[r] * __expf(mx8[r] - v1) + 1.f; mx8[r] = v1; }
            else             { sm8[r] += __expf(v1 - mx8[r]); }
        }
    }
#pragma unroll
    for (int r = 0; r < 8; r++) {
        rmax[wave][r + 8 * hv][n] = mx8[r];
        rsum[wave][r + 8 * hv][n] = sm8[r];
    }
    __syncthreads();
    if (tid < 16) {
        float M = -1e30f, S = 0.f;
        for (int w = 0; w < 8; w++)
            for (int c = 0; c < 16; c++) {
                float m2 = rmax[w][tid][c], s2 = rsum[w][tid][c];
                if (m2 > M) { S = S * __expf(M - m2) + s2; M = m2; }
                else        { S += s2 * __expf(m2 - M); }
            }
        float lse = M + __logf(S);
        int b = (m0 & 63) + tid;
        int g = gcol[tid];
        glp[(long)tt * BATCH + b] = (g != PAD_ID) ? (gval[tid] - lse) : 0.f;
    }
}

__global__ void k_reduce(const float* __restrict__ glp, float* __restrict__ out) {
    int b = threadIdx.x;
    if (b < BATCH) {
        float s = 0.f;
        for (int t = 0; t < T_STEPS; t++) s += glp[(long)t * BATCH + b];
        out[b] = s;
    }
}

// ---------------- host ----------------
extern "C" void kernel_launch(void* const* d_in, const int* in_sizes, int n_in,
                              void* d_out, int out_size, void* d_ws, size_t ws_size,
                              hipStream_t stream) {
    const int*   ids  = (const int*)  d_in[0];
    const float* vc   = (const float*)d_in[1];
    const float* h0   = (const float*)d_in[2];
    const float* c0   = (const float*)d_in[3];
    const float* Et   = (const float*)d_in[4];
    const float* Watt = (const float*)d_in[5];
    const float* batt = (const float*)d_in[6];
    const float* Wih  = (const float*)d_in[7];
    const float* Whh  = (const float*)d_in[8];
    const float* Wu   = (const float*)d_in[9];
    const float* Wv   = (const float*)d_in[10];
    const float* bvoc = (const float*)d_in[11];

    char* ws = (char*)d_ws;
    size_t off = 0;
    auto take = [&](size_t bytes) -> char* {
        char* p = ws + off;
        off = (off + bytes + 255) & ~(size_t)255;
        return p;
    };
    bf16*  Watt_b = (bf16*) take((size_t)HID * 2 * HID * 2);
    bf16*  vc_b   = (bf16*) take((size_t)BATCH * SEQ * 2 * HID * 2);
    bf16*  encp_b = (bf16*) take((size_t)BATCH * SEQ * HID * 2);
    bf16*  Wc_b   = (bf16*) take((size_t)4 * HID * KX * 2);
    bf16*  Wu_b   = (bf16*) take((size_t)HID * KU * 2);
    bf16*  Wv_b   = (bf16*) take((size_t)VOCAB * HID * 2);
    bf16*  Y_b    = (bf16*) take((size_t)T_STEPS * BATCH * EMB * 2);
    bf16*  X0     = (bf16*) take((size_t)BATCH * KX * 2);
    bf16*  X1     = (bf16*) take((size_t)BATCH * KX * 2);
    bf16*  Xu_b   = (bf16*) take((size_t)BATCH * KU * 2);
    float* h_f    = (float*)take((size_t)BATCH * HID * 4);
    float* c_f    = (float*)take((size_t)BATCH * HID * 4);
    bf16*  comb_b = (bf16*) take((size_t)T_STEPS * BATCH * HID * 2);
    float* glp    = (float*)take((size_t)T_STEPS * BATCH * 4);

    // prep: bf16 conversions + weight packing + embedding gather + state init
    long n;
    n = (long)HID * 2 * HID;
    k_f2bf<<<dim3((unsigned)((n + 255) / 256)), 256, 0, stream>>>(Watt, Watt_b, n);
    n = (long)BATCH * SEQ * 2 * HID;
    k_f2bf<<<dim3((unsigned)((n + 255) / 256)), 256, 0, stream>>>(vc, vc_b, n);
    n = (long)HID * KU;
    k_f2bf<<<dim3((unsigned)((n + 255) / 256)), 256, 0, stream>>>(Wu, Wu_b, n);
    n = (long)VOCAB * HID;
    k_f2bf<<<dim3((unsigned)((n + 255) / 256)), 256, 0, stream>>>(Wv, Wv_b, n);
    n = (long)4 * HID * KX;
    k_wcomb<<<dim3((unsigned)((n + 255) / 256)), 256, 0, stream>>>(Wih, Whh, Wc_b, n);
    n = (long)T_STEPS * BATCH * EMB;
    k_embed<<<dim3((unsigned)((n + 255) / 256)), 256, 0, stream>>>(ids, Et, Y_b, n);
    k_init<<<dim3((BATCH * KX + 255) / 256), 256, 0, stream>>>(h0, c0, Y_b, X0, h_f, c_f);

    // enc_proj GEMM
    k_encproj<<<dim3(BATCH * SEQ / 16, HID / 128), 128, 0, stream>>>(vc_b, Watt_b, batt, encp_b);

    // sequential scan (ping-pong X to avoid cross-block h RAW hazard between steps)
    for (int t = 0; t < T_STEPS; t++) {
        bf16* Xc = (t & 1) ? X1 : X0;
        bf16* Xn = (t & 1) ? X0 : X1;
        k_step1<<<dim3(BATCH / 16, HID / 32), 128, 0, stream>>>(Xc, Wc_b, c_f, h_f, Xn);
        k_step2<<<dim3(BATCH), 256, 0, stream>>>(h_f, encp_b, vc_b, Xu_b);
        k_step3<<<dim3(BATCH / 16, HID / 128), 128, 0, stream>>>(Xu_b, Wu_b, Xn, comb_b, Y_b, t);
    }

    // vocab scoring with online logsumexp + deterministic per-batch sum
    k_score<<<dim3(T_STEPS * BATCH / 16), 256, 0, stream>>>(comb_b, Wv_b, bvoc, ids, glp);
    k_reduce<<<dim3(1), 64, 0, stream>>>(glp, (float*)d_out);
}
